// PINNS_19241453486649
// MI455X (gfx1250) — matrix operand order repaired and checked by reading, not compiled, not run on hardware
//
#include <hip/hip_runtime.h>
#include <stdint.h>
#include <stddef.h>

#define SEQ_LEN 256
#define BATCH   512
#define HIDDEN  512
#define FOURH   2048

typedef __attribute__((ext_vector_type(16))) __bf16 bf16x16;
typedef __attribute__((ext_vector_type(8)))  __bf16 bf16x8;
typedef __attribute__((ext_vector_type(8)))  float  f32x8;
typedef __attribute__((ext_vector_type(4)))  unsigned int u32x4;

// ---------------- device helpers ----------------

static __device__ __forceinline__ unsigned short f2bf(float f) {
  union { float f; uint32_t u; } c; c.f = f;
  uint32_t u = c.u;
  uint32_t r = (u + 0x7FFFu + ((u >> 16) & 1u)) >> 16;  // round-to-nearest-even
  return (unsigned short)r;
}
static __device__ __forceinline__ float sigm(float x) {
  return 1.0f / (1.0f + __expf(-x));
}
static __device__ __forceinline__ float tanh_fast(float x) {
  return 1.0f - 2.0f / (__expf(2.0f * x) + 1.0f);
}

// ---------------- prep kernels ----------------

// Build bf16 weights: W0bf = bf16(Whh0) [2048][512]; W1bf [2048][1024] = bf16([Wih1 | Whh1])
__global__ void prep_w_kernel(const float* __restrict__ Whh0,
                              const float* __restrict__ Wih1,
                              const float* __restrict__ Whh1,
                              unsigned short* __restrict__ W0bf,
                              unsigned short* __restrict__ W1bf) {
  int idx = blockIdx.x * blockDim.x + threadIdx.x;   // 0 .. 2048*1024-1
  int n = idx >> 10;
  int k = idx & 1023;
  float v = (k < 512) ? Wih1[n * 512 + k] : Whh1[n * 512 + (k - 512)];
  W1bf[idx] = f2bf(v);
  if (idx < FOURH * HIDDEN) W0bf[idx] = f2bf(Whh0[idx]);
}

// Bias sums + initial h/c state
__global__ void prep_state_kernel(const float* __restrict__ h0,
                                  const float* __restrict__ c0,
                                  const float* __restrict__ bih0, const float* __restrict__ bhh0,
                                  const float* __restrict__ bih1, const float* __restrict__ bhh1,
                                  unsigned short* __restrict__ Hb0,
                                  unsigned short* __restrict__ Hb1,
                                  float* __restrict__ C0, float* __restrict__ C1,
                                  float* __restrict__ bias0, float* __restrict__ bias1) {
  int i = blockIdx.x * blockDim.x + threadIdx.x;  // 0 .. 262143
  if (i < FOURH) {
    bias0[i] = bih0[i] + bhh0[i];
    bias1[i] = bih1[i] + bhh1[i];
  }
  Hb0[i] = f2bf(h0[i]);
  Hb1[i] = f2bf(h0[BATCH * HIDDEN + i]);
  C0[i] = c0[i];
  C1[i] = c0[BATCH * HIDDEN + i];
}

// ---------------- fused GEMM + LSTM cell, one time step, one layer ----------------
// gates[b][n] = sum_k A[b][k] * W[n][k]  (+ bias[n] + x[t][b] . Wx[n][:3])
// A split: K chunks 0..15 from A0, 16..31 from A1 (each [512][512] bf16).
// Block: 128 threads (4 waves). Tile: 64 batch rows x (4 gates x 32 j cols).
// LDS ping-pong software pipeline: global loads for chunk c+1 are issued before
// the WMMAs of chunk c; one barrier per chunk.
__global__ __launch_bounds__(128)
void lstm_step_kernel(const unsigned short* __restrict__ A0,
                      const unsigned short* __restrict__ A1,
                      int kChunks,
                      const unsigned short* __restrict__ W, int Kw,
                      const float* __restrict__ bias,
                      const float* __restrict__ xt,   // x + t*B*3, or null
                      const float* __restrict__ Wx,   // Wih0 [2048][3], or null
                      float* __restrict__ C,
                      unsigned short* __restrict__ Hout,
                      float* __restrict__ Hf32) {
  __shared__ __align__(16) unsigned short ldsA[2][64 * 40];    // 64 rows x 32 K (+8 pad)
  __shared__ __align__(16) unsigned short ldsB[2][128 * 40];   // 128 W-rows x 32 K (+8 pad)

  const int tid   = threadIdx.x;
  const int mBase = blockIdx.x * 64;
  const int jBase = blockIdx.y * 32;
  const int wave  = tid >> 5;
  const int lane  = tid & 31;
  const int waveM = wave & 1;
  const int waveJ = wave >> 1;
  const int lr    = lane & 15;   // row (A/M) or col (B/N, C/N) within frag
  const int lh    = lane >> 4;   // half selector

  // per-thread staging addresses (rows/segments fixed, only K offset varies)
  const int rowA = (tid >> 2), segA = (tid & 3);               // + i*32 rows
  const int rowB = (tid >> 2), segB = (tid & 3);               // + i*32 rows

  f32x8 acc[2][4];
  const f32x8 zero = {0.f, 0.f, 0.f, 0.f, 0.f, 0.f, 0.f, 0.f};
#pragma unroll
  for (int mi = 0; mi < 2; ++mi)
#pragma unroll
    for (int g = 0; g < 4; ++g) acc[mi][g] = zero;

  u32x4 ra[2], rb[4];

  auto loadTiles = [&](int c) {
    const unsigned short* Asrc = (c < 16) ? A0 : A1;
    const int kA = (c & 15) * 32;
    const int kW = c * 32;
#pragma unroll
    for (int i = 0; i < 2; ++i) {
      int row = rowA + i * 32;
      ra[i] = *(const u32x4*)(Asrc + (size_t)(mBase + row) * HIDDEN + kA + segA * 8);
    }
#pragma unroll
    for (int i = 0; i < 4; ++i) {
      int rr = rowB + i * 32;
      int n = (rr >> 5) * HIDDEN + jBase + (rr & 31);          // gate*512 + j
      rb[i] = *(const u32x4*)(W + (size_t)n * Kw + kW + segB * 8);
    }
  };
  auto storeTiles = [&](int p) {
#pragma unroll
    for (int i = 0; i < 2; ++i)
      *(u32x4*)&ldsA[p][(rowA + i * 32) * 40 + segA * 8] = ra[i];
#pragma unroll
    for (int i = 0; i < 4; ++i)
      *(u32x4*)&ldsB[p][(rowB + i * 32) * 40 + segB * 8] = rb[i];
  };

  // prologue: chunk 0 into buffer 0
  loadTiles(0);
  storeTiles(0);
  __syncthreads();

  for (int c = 0; c < kChunks; ++c) {
    const int p = c & 1;
    const bool more = (c + 1 < kChunks);
    if (more) loadTiles(c + 1);   // overlap global latency with WMMAs below

    // A fragments (16-bit A 16x32 layout: lane<16 -> K{0..7,16..23}; lane>=16 -> +8)
    bf16x16 afr[2];
#pragma unroll
    for (int mi = 0; mi < 2; ++mi) {
      const unsigned short* pa = &ldsA[p][(waveM * 32 + mi * 16 + lr) * 40];
      bf16x8 lo = *(const bf16x8*)(pa + lh * 8);
      bf16x8 hi = *(const bf16x8*)(pa + 16 + lh * 8);
      afr[mi] = __builtin_shufflevector(lo, hi, 0, 1, 2, 3, 4, 5, 6, 7,
                                        8, 9, 10, 11, 12, 13, 14, 15);
    }
    // B fragments (16-bit B 32x16: lane<16 -> K{0..7,16..23}; lane>=16 -> +8; col = lane&15)
    bf16x16 bfr[4];
#pragma unroll
    for (int g = 0; g < 4; ++g) {
      const unsigned short* pb = &ldsB[p][(g * 32 + waveJ * 16 + lr) * 40];
      bf16x8 lo = *(const bf16x8*)(pb + lh * 8);
      bf16x8 hi = *(const bf16x8*)(pb + 16 + lh * 8);
      bfr[g] = __builtin_shufflevector(lo, hi, 0, 1, 2, 3, 4, 5, 6, 7,
                                       8, 9, 10, 11, 12, 13, 14, 15);
    }
#pragma unroll
    for (int mi = 0; mi < 2; ++mi)
#pragma unroll
      for (int g = 0; g < 4; ++g)
        acc[mi][g] = __builtin_amdgcn_wmma_f32_16x16x32_bf16(
            false, afr[mi], false, bfr[g], (short)0, acc[mi][g], false, false);

    if (more) {
      storeTiles(1 - p);   // waits on its own loads by data dependence
      __syncthreads();     // stores visible before next chunk's frag reads
    }
  }

  // ---------------- LSTM cell epilogue (register-local: all 4 gates per (b,j)) ----------
  const int jg = jBase + waveJ * 16 + lr;   // global j in [0,512)
  const float bI = bias[jg];
  const float bF = bias[HIDDEN + jg];
  const float bG = bias[2 * HIDDEN + jg];
  const float bO = bias[3 * HIDDEN + jg];

  float wx[4][3];
  if (Wx) {
#pragma unroll
    for (int g = 0; g < 4; ++g)
#pragma unroll
      for (int k = 0; k < 3; ++k) wx[g][k] = Wx[(g * HIDDEN + jg) * 3 + k];
  }

#pragma unroll
  for (int mi = 0; mi < 2; ++mi) {
#pragma unroll
    for (int r = 0; r < 8; ++r) {
      int b = mBase + waveM * 32 + mi * 16 + lh * 8 + r;  // C frag: M = 8*half + vgpr
      float gi = acc[mi][0][r] + bI;
      float gf = acc[mi][1][r] + bF;
      float gg = acc[mi][2][r] + bG;
      float go = acc[mi][3][r] + bO;
      if (xt) {
        float x0 = xt[b * 3 + 0], x1 = xt[b * 3 + 1], x2 = xt[b * 3 + 2];
        gi += x0 * wx[0][0] + x1 * wx[0][1] + x2 * wx[0][2];
        gf += x0 * wx[1][0] + x1 * wx[1][1] + x2 * wx[1][2];
        gg += x0 * wx[2][0] + x1 * wx[2][1] + x2 * wx[2][2];
        go += x0 * wx[3][0] + x1 * wx[3][1] + x2 * wx[3][2];
      }
      size_t idx = (size_t)b * HIDDEN + jg;
      float cOld = C[idx];
      float iv = sigm(gi), fv = sigm(gf), gv = tanh_fast(gg), ov = sigm(go);
      float cN = fv * cOld + iv * gv;
      float h = ov * tanh_fast(cN);
      C[idx] = cN;
      Hf32[idx] = h;
      Hout[idx] = f2bf(h);
    }
  }
}

// ---------------- head ----------------

// td[b][n] = [h0T | h1T] . Wtd[n] + btd[n];  b=blockIdx (512), n=threadIdx (256)
__global__ void td_kernel(const float* __restrict__ H0, const float* __restrict__ H1,
                          const float* __restrict__ Wtd, const float* __restrict__ btd,
                          float* __restrict__ td) {
  int b = blockIdx.x, n = threadIdx.x;
  const float* wr = Wtd + (size_t)n * (2 * HIDDEN);
  float acc = btd[n];
  const float* h0 = H0 + (size_t)b * HIDDEN;
  const float* h1 = H1 + (size_t)b * HIDDEN;
  for (int k = 0; k < HIDDEN; ++k) acc += h0[k] * wr[k];
  for (int k = 0; k < HIDDEN; ++k) acc += h1[k] * wr[HIDDEN + k];
  td[(size_t)b * SEQ_LEN + n] = acc;
}

// out[b][o] = relu(td[b] @ Wfc1.T + bfc1) @ Wfc2.T + bfc2
__global__ void fc_kernel(const float* __restrict__ td,
                          const float* __restrict__ Wfc1, const float* __restrict__ bfc1,
                          const float* __restrict__ Wfc2, const float* __restrict__ bfc2,
                          float* __restrict__ out) {
  int b = blockIdx.x * blockDim.x + threadIdx.x;
  if (b >= BATCH) return;
  const float* t = td + (size_t)b * SEQ_LEN;
  float f1[20];
  for (int j = 0; j < 20; ++j) {
    float s = bfc1[j];
    const float* w = Wfc1 + (size_t)j * SEQ_LEN;
    for (int n = 0; n < SEQ_LEN; ++n) s += t[n] * w[n];
    f1[j] = fmaxf(s, 0.0f);
  }
  for (int o = 0; o < 2; ++o) {
    float s = bfc2[o];
    for (int j = 0; j < 20; ++j) s += f1[j] * Wfc2[o * 20 + j];
    out[(size_t)b * 2 + o] = s;
  }
}

// ---------------- workspace layout (bytes) ----------------
#define OFF_W0   ((size_t)0)              // 2048*512*2   = 2,097,152
#define OFF_W1   ((size_t)2097152)        // 2048*1024*2  = 4,194,304
#define OFF_B0   ((size_t)6291456)        // 8192
#define OFF_B1   ((size_t)6299648)        // 8192
#define OFF_H0   ((size_t)6307840)        // 2 * 524,288 bf16 h0 double buffer
#define OFF_H1   ((size_t)7356416)        // 2 * 524,288
#define OFF_C0   ((size_t)8404992)        // 1,048,576
#define OFF_C1   ((size_t)9453568)        // 1,048,576
#define OFF_HF0  ((size_t)10502144)       // 1,048,576
#define OFF_HF1  ((size_t)11550720)       // 1,048,576
#define OFF_TD   ((size_t)12599296)       // 524,288
// total ~13.1 MB

extern "C" void kernel_launch(void* const* d_in, const int* in_sizes, int n_in,
                              void* d_out, int out_size, void* d_ws, size_t ws_size,
                              hipStream_t stream) {
  (void)in_sizes; (void)n_in; (void)out_size; (void)ws_size;
  const float* x    = (const float*)d_in[0];
  const float* h0   = (const float*)d_in[1];
  const float* c0   = (const float*)d_in[2];
  const float* Wih0 = (const float*)d_in[3];
  const float* Whh0 = (const float*)d_in[4];
  const float* bih0 = (const float*)d_in[5];
  const float* bhh0 = (const float*)d_in[6];
  const float* Wih1 = (const float*)d_in[7];
  const float* Whh1 = (const float*)d_in[8];
  const float* bih1 = (const float*)d_in[9];
  const float* bhh1 = (const float*)d_in[10];
  const float* Wtd  = (const float*)d_in[11];
  const float* btd  = (const float*)d_in[12];
  const float* Wfc1 = (const float*)d_in[13];
  const float* bfc1 = (const float*)d_in[14];
  const float* Wfc2 = (const float*)d_in[15];
  const float* bfc2 = (const float*)d_in[16];
  float* out = (float*)d_out;

  char* ws = (char*)d_ws;
  unsigned short* W0bf = (unsigned short*)(ws + OFF_W0);
  unsigned short* W1bf = (unsigned short*)(ws + OFF_W1);
  float* bias0 = (float*)(ws + OFF_B0);
  float* bias1 = (float*)(ws + OFF_B1);
  unsigned short* Hb0[2] = {(unsigned short*)(ws + OFF_H0),
                            (unsigned short*)(ws + OFF_H0 + 524288)};
  unsigned short* Hb1[2] = {(unsigned short*)(ws + OFF_H1),
                            (unsigned short*)(ws + OFF_H1 + 524288)};
  float* C0   = (float*)(ws + OFF_C0);
  float* C1   = (float*)(ws + OFF_C1);
  float* HF0  = (float*)(ws + OFF_HF0);
  float* HF1  = (float*)(ws + OFF_HF1);
  float* td   = (float*)(ws + OFF_TD);

  // prep: weights (2048*1024 threads), state (512*512 threads)
  prep_w_kernel<<<8192, 256, 0, stream>>>(Whh0, Wih1, Whh1, W0bf, W1bf);
  prep_state_kernel<<<1024, 256, 0, stream>>>(h0, c0, bih0, bhh0, bih1, bhh1,
                                              Hb0[0], Hb1[0], C0, C1, bias0, bias1);

  dim3 gemmGrid(BATCH / 64, HIDDEN / 32);  // 8 x 16 = 128 blocks
  for (int t = 0; t < SEQ_LEN; ++t) {
    int p = t & 1;
    const float* xt = x + (size_t)t * BATCH * 3;
    // layer 0: A = h0(prev), K=512; + x.Wih0 epilogue; writes h0(next)
    lstm_step_kernel<<<gemmGrid, 128, 0, stream>>>(
        Hb0[p], Hb0[p], 16, W0bf, HIDDEN, bias0, xt, Wih0, C0, Hb0[1 - p], HF0);
    // layer 1: A = [y0_t | h1(prev)], K=1024; writes h1(next)
    lstm_step_kernel<<<gemmGrid, 128, 0, stream>>>(
        Hb0[1 - p], Hb1[p], 32, W1bf, 2 * HIDDEN, bias1, nullptr, nullptr,
        C1, Hb1[1 - p], HF1);
  }

  td_kernel<<<BATCH, SEQ_LEN, 0, stream>>>(HF0, HF1, Wtd, btd, td);
  fc_kernel<<<2, 256, 0, stream>>>(td, Wfc1, bfc1, Wfc2, bfc2, out);
}
